// GT_73065983640293
// MI455X (gfx1250) — compile-verified, matrix operand order checked by reading, not run on hardware
//
#include <hip/hip_runtime.h>
#include <hip/hip_bf16.h>
#include <math.h>

#define N_NODES 20000
#define N_EDGES 640000
#define IN_DIM  256
#define DMODEL  128
#define NHEAD   8
#define IDIMV   64
#define NLAYERS 10

typedef __bf16 bf16;
typedef __attribute__((ext_vector_type(16))) __bf16 v16bf;
typedef __attribute__((ext_vector_type(8)))  __bf16 v8bf;
typedef __attribute__((ext_vector_type(4)))  __bf16 v4bf;
typedef __attribute__((ext_vector_type(8)))  float  v8f;

// ---------------------------------------------------------------------------
// fp32 atomic add, no-return form -> native global_atomic_add_f32 (STOREcnt)
// ---------------------------------------------------------------------------
__device__ __forceinline__ void atomic_add_f32_nr(float* p, float v) {
  asm volatile("global_atomic_add_f32 %0, %1, off" :: "v"(p), "v"(v) : "memory");
}

// ---------------------------------------------------------------------------
// fp32 -> bf16 conversion
// ---------------------------------------------------------------------------
__global__ void k_f32_to_bf16(const float* __restrict__ in, bf16* __restrict__ out, int n) {
  int i = blockIdx.x * blockDim.x + threadIdx.x;
  if (i < n) out[i] = (bf16)in[i];
}

__global__ void k_zero_f32(float* __restrict__ p, int n) {
  int i = blockIdx.x * blockDim.x + threadIdx.x;
  if (i < n) p[i] = 0.0f;
}

__global__ void k_copy_f32(const float* __restrict__ in, float* __restrict__ out, int n) {
  int i = blockIdx.x * blockDim.x + threadIdx.x;
  if (i < n) out[i] = in[i];
}

// ---------------------------------------------------------------------------
// WMMA bf16 GEMM:  C[M,DOUT] = A[M,K] * B[DOUT,K]^T + bias, optional act.
// A-operand fragment layout (16-bit, 16x32):
//   lane L: row = L%16, half = L/16
//   frag elems 0..7  = K[half*8 + 0..7]        (one 16B load)
//   frag elems 8..15 = K[16 + half*8 + 0..7]   (one 16B load)
// B rows of W (row-major [out,in]) load with the same pattern -> B columns.
// ---------------------------------------------------------------------------
__device__ __forceinline__ v16bf load_frag(const bf16* __restrict__ base, int ld,
                                           int row, int k0, int half) {
  const bf16* p = base + (size_t)row * ld + k0 + half * 8;
  v8bf lo = *(const v8bf*)(p);
  v8bf hi = *(const v8bf*)(p + 16);
  v16bf r;
#pragma unroll
  for (int i = 0; i < 8; ++i) { r[i] = lo[i]; r[8 + i] = hi[i]; }
  return r;
}

// ACT: 0 = none, 1 = relu, 2 = selu
template <int K, int DOUT, int ACT>
__global__ void k_gemm_bf16(const bf16* __restrict__ A, const bf16* __restrict__ B,
                            const float* __restrict__ bias,
                            float* __restrict__ Cf, bf16* __restrict__ Cb, int M) {
  constexpr int NT = DOUT / 16;
  const int wave = threadIdx.x >> 5;
  const int lane = threadIdx.x & 31;
  const int tile = blockIdx.x * (blockDim.x >> 5) + wave;
  const int tm = tile / NT;
  const int tn = tile - tm * NT;
  if (tm * 16 >= M) return;
  const int half = lane >> 4;
  const int lr   = lane & 15;
  const int arow = tm * 16 + lr;
  const int bcol = tn * 16 + lr;

  v8f acc = {};
#pragma unroll
  for (int k0 = 0; k0 < K; k0 += 32) {
    v16bf a = load_frag(A, K, arow, k0, half);
    v16bf b = load_frag(B, K, bcol, k0, half);
    acc = __builtin_amdgcn_wmma_f32_16x16x32_bf16(false, a, false, b,
                                                  (short)0, acc, false, false);
  }

  // C layout: lane L -> n = L%16 ; vgpr j -> m = tm*16 + j + 8*(L/16)
  const int n  = tn * 16 + lr;
  const float bs = bias ? bias[n] : 0.0f;
#pragma unroll
  for (int j = 0; j < 8; ++j) {
    const int m = tm * 16 + j + half * 8;
    float v = acc[j] + bs;
    if (ACT == 1) v = fmaxf(v, 0.0f);
    else if (ACT == 2) v = (v > 0.0f) ? 1.0507009873554805f * v
                                      : 1.7580993408473766f * (expf(v) - 1.0f);
    if (Cf) Cf[(size_t)m * DOUT + n] = v;
    if (Cb) Cb[(size_t)m * DOUT + n] = (bf16)v;
  }
}

// ---------------------------------------------------------------------------
// Edge attention: one wave32 per edge. lane covers channels [4*lane, 4*lane+3]
// head h = lane/4 (HD=16 = 4 lanes). Per-head dot via 2x shfl_xor.
// score = exp(clamp(dot/4, -10, 10)) * 0.5 ; scatter V*score and score.
// ---------------------------------------------------------------------------
__global__ void k_edge_attn(const int* __restrict__ src, const int* __restrict__ dst,
                            const bf16* __restrict__ Q, const bf16* __restrict__ Kb,
                            const bf16* __restrict__ V,
                            float* __restrict__ wV, float* __restrict__ zn) {
  const int gid  = blockIdx.x * blockDim.x + threadIdx.x;
  const int e    = gid >> 5;
  if (e >= N_EDGES) return;
  const int lane = gid & 31;
  const int s = src[e];
  const int d = dst[e];

  v4bf kv = *(const v4bf*)(Kb + (size_t)s * DMODEL + lane * 4);
  v4bf qv = *(const v4bf*)(Q  + (size_t)d * DMODEL + lane * 4);
  float dot = 0.0f;
#pragma unroll
  for (int j = 0; j < 4; ++j) dot += (float)kv[j] * (float)qv[j];
  dot += __shfl_xor(dot, 1, 32);
  dot += __shfl_xor(dot, 2, 32);            // quartet now holds full head dot

  float sc = dot * 0.25f;                   // / sqrt(HD=16)
  sc = fminf(fmaxf(sc, -10.0f), 10.0f);
  sc = __expf(sc) * 0.5f;

  v4bf vv = *(const v4bf*)(V + (size_t)s * DMODEL + lane * 4);
  float* wrow = wV + (size_t)d * DMODEL + lane * 4;
#pragma unroll
  for (int j = 0; j < 4; ++j) atomic_add_f32_nr(wrow + j, (float)vv[j] * sc);
  if ((lane & 3) == 0) atomic_add_f32_nr(zn + (size_t)d * NHEAD + (lane >> 2), sc);
}

// attn = wV/(zn+1e-6) + IWi   -> bf16 (input to Wo GEMM)
__global__ void k_attn_combine(const float* __restrict__ wV, const float* __restrict__ zn,
                               const float* __restrict__ IWi, bf16* __restrict__ Ab) {
  const int row = blockIdx.x;
  const int lane = threadIdx.x;
  const int e0 = lane * 4;
  const float inv = 1.0f / (zn[(size_t)row * NHEAD + (e0 >> 4)] + 1e-6f);
  float4 wv = *(const float4*)(wV  + (size_t)row * DMODEL + e0);
  float4 iw = *(const float4*)(IWi + (size_t)row * DMODEL + e0);
  v4bf o;
  o[0] = (bf16)(wv.x * inv + iw.x);
  o[1] = (bf16)(wv.y * inv + iw.y);
  o[2] = (bf16)(wv.z * inv + iw.z);
  o[3] = (bf16)(wv.w * inv + iw.w);
  *(v4bf*)(Ab + (size_t)row * DMODEL + e0) = o;
}

// out = LayerNorm(X + Y) * g + b  -> fp32 (residual path) and bf16 (GEMM path)
__global__ void k_add_ln(const float* __restrict__ X, const float* __restrict__ Y,
                         const float* __restrict__ g, const float* __restrict__ b,
                         float* __restrict__ Of, bf16* __restrict__ Ob) {
  const int row = blockIdx.x;
  const int lane = threadIdx.x;
  float4 x = *(const float4*)(X + (size_t)row * DMODEL + lane * 4);
  float4 y = *(const float4*)(Y + (size_t)row * DMODEL + lane * 4);
  float v0 = x.x + y.x, v1 = x.y + y.y, v2 = x.z + y.z, v3 = x.w + y.w;

  float s = v0 + v1 + v2 + v3;
#pragma unroll
  for (int off = 16; off; off >>= 1) s += __shfl_xor(s, off, 32);
  const float mean = s * (1.0f / 128.0f);
  const float d0 = v0 - mean, d1 = v1 - mean, d2 = v2 - mean, d3 = v3 - mean;
  float q = d0 * d0 + d1 * d1 + d2 * d2 + d3 * d3;
#pragma unroll
  for (int off = 16; off; off >>= 1) q += __shfl_xor(q, off, 32);
  const float r = rsqrtf(q * (1.0f / 128.0f) + 1e-5f);

  float4 gg = *(const float4*)(g + lane * 4);
  float4 bb = *(const float4*)(b + lane * 4);
  const float o0 = d0 * r * gg.x + bb.x;
  const float o1 = d1 * r * gg.y + bb.y;
  const float o2 = d2 * r * gg.z + bb.z;
  const float o3 = d3 * r * gg.w + bb.w;
  if (Of) { float4 of = {o0, o1, o2, o3};
            *(float4*)(Of + (size_t)row * DMODEL + lane * 4) = of; }
  if (Ob) { v4bf ob; ob[0]=(bf16)o0; ob[1]=(bf16)o1; ob[2]=(bf16)o2; ob[3]=(bf16)o3;
            *(v4bf*)(Ob + (size_t)row * DMODEL + lane * 4) = ob; }
}

// ---------------------------------------------------------------------------
// Host-side orchestration
// ---------------------------------------------------------------------------
extern "C" void kernel_launch(void* const* d_in, const int* in_sizes, int n_in,
                              void* d_out, int out_size, void* d_ws, size_t ws_size,
                              hipStream_t stream) {
  (void)in_sizes; (void)n_in; (void)out_size; (void)ws_size;
  const float* x     = (const float*)d_in[0];
  const float* I     = (const float*)d_in[1];
  const int*   src   = (const int*)d_in[2];
  const int*   dst   = (const int*)d_in[3];
  const float* W_emb = (const float*)d_in[4];
  const float* Wq = (const float*)d_in[5],  *bq = (const float*)d_in[6];
  const float* Wk = (const float*)d_in[7],  *bk = (const float*)d_in[8];
  const float* Wv = (const float*)d_in[9],  *bv = (const float*)d_in[10];
  const float* Wi = (const float*)d_in[11], *bi = (const float*)d_in[12];
  const float* Wo = (const float*)d_in[13], *bo = (const float*)d_in[14];
  const float* ln1_g = (const float*)d_in[15], *ln1_b = (const float*)d_in[16];
  const float* ln2_g = (const float*)d_in[17], *ln2_b = (const float*)d_in[18];
  const float* W_ff1 = (const float*)d_in[19], *b_ff1 = (const float*)d_in[20];
  const float* W_ff2 = (const float*)d_in[21], *b_ff2 = (const float*)d_in[22];
  const float* W_m1  = (const float*)d_in[23], *b_m1  = (const float*)d_in[24];
  const float* W_m2  = (const float*)d_in[25], *b_m2  = (const float*)d_in[26];
  float* out_h = (float*)d_out;                       // [N,128]
  float* out_x = (float*)d_out + (size_t)N_NODES * DMODEL;  // [N,256]

  // ---- workspace carve-up (256B aligned) ----
  size_t off = 0;
  auto alloc = [&](size_t bytes) -> char* {
    char* p = (char*)d_ws + off;
    off += (bytes + 255) & ~(size_t)255;
    return p;
  };
  bf16* wb_emb = (bf16*)alloc(DMODEL * IN_DIM * 2);
  bf16* wb_q   = (bf16*)alloc(DMODEL * DMODEL * 2);
  bf16* wb_k   = (bf16*)alloc(DMODEL * DMODEL * 2);
  bf16* wb_v   = (bf16*)alloc(DMODEL * DMODEL * 2);
  bf16* wb_o   = (bf16*)alloc(DMODEL * DMODEL * 2);
  bf16* wb_i   = (bf16*)alloc(DMODEL * IDIMV * 2);
  bf16* wb_ff1 = (bf16*)alloc(2 * DMODEL * DMODEL * 2);
  bf16* wb_ff2 = (bf16*)alloc(DMODEL * 2 * DMODEL * 2);
  bf16* wb_m1  = (bf16*)alloc(DMODEL * DMODEL * 2);
  bf16* wb_m2  = (bf16*)alloc(IN_DIM * DMODEL * 2);
  bf16* x_bf   = (bf16*)alloc((size_t)N_NODES * IN_DIM * 2);   // reused as f1_bf
  bf16* i_bf   = (bf16*)alloc((size_t)N_NODES * IDIMV * 2);
  float* h_f   = (float*)alloc((size_t)N_NODES * DMODEL * 4);
  bf16* h_bf   = (bf16*)alloc((size_t)N_NODES * DMODEL * 2);
  float* iwi_f = (float*)alloc((size_t)N_NODES * DMODEL * 4);
  bf16* q_bf   = (bf16*)alloc((size_t)N_NODES * DMODEL * 2);
  bf16* k_bf   = (bf16*)alloc((size_t)N_NODES * DMODEL * 2);
  bf16* v_bf   = (bf16*)alloc((size_t)N_NODES * DMODEL * 2);
  float* wv_f  = (float*)alloc((size_t)N_NODES * DMODEL * 4);  // reused as o_f / f2_f
  float* zn_f  = (float*)alloc((size_t)N_NODES * NHEAD * 4);
  bf16* a_bf   = (bf16*)alloc((size_t)N_NODES * DMODEL * 2);   // reused as m_bf
  float* t_f   = (float*)alloc((size_t)N_NODES * DMODEL * 4);
  bf16* t_bf   = (bf16*)alloc((size_t)N_NODES * DMODEL * 2);
  bf16* f1_bf  = x_bf;

  auto cv = [&](const float* in, bf16* out, int n) {
    k_f32_to_bf16<<<(n + 255) / 256, 256, 0, stream>>>(in, out, n);
  };
  auto ggrid = [&](int dout) {  // 8 waves / block, one 16x16 tile per wave
    int tiles = (N_NODES / 16) * (dout / 16);
    return dim3((tiles + 7) / 8);
  };
  const dim3 B256(256), B32(32);

  // ---- one-time conversions ----
  cv(x, x_bf, N_NODES * IN_DIM);
  cv(I, i_bf, N_NODES * IDIMV);
  cv(W_emb, wb_emb, DMODEL * IN_DIM);
  cv(Wq, wb_q, DMODEL * DMODEL);  cv(Wk, wb_k, DMODEL * DMODEL);
  cv(Wv, wb_v, DMODEL * DMODEL);  cv(Wo, wb_o, DMODEL * DMODEL);
  cv(Wi, wb_i, DMODEL * IDIMV);
  cv(W_ff1, wb_ff1, 2 * DMODEL * DMODEL);
  cv(W_ff2, wb_ff2, DMODEL * 2 * DMODEL);
  cv(W_m1, wb_m1, DMODEL * DMODEL);
  cv(W_m2, wb_m2, IN_DIM * DMODEL);

  // h0 = x @ W_emb.T    |    IWi = I @ Wi.T + bi  (loop-invariant)
  k_gemm_bf16<IN_DIM, DMODEL, 0><<<ggrid(DMODEL), B256, 0, stream>>>(
      x_bf, wb_emb, nullptr, h_f, h_bf, N_NODES);
  k_gemm_bf16<IDIMV, DMODEL, 0><<<ggrid(DMODEL), B256, 0, stream>>>(
      i_bf, wb_i, bi, iwi_f, nullptr, N_NODES);

  for (int layer = 0; layer < NLAYERS; ++layer) {
    k_gemm_bf16<DMODEL, DMODEL, 0><<<ggrid(DMODEL), B256, 0, stream>>>(
        h_bf, wb_q, bq, nullptr, q_bf, N_NODES);
    k_gemm_bf16<DMODEL, DMODEL, 0><<<ggrid(DMODEL), B256, 0, stream>>>(
        h_bf, wb_k, bk, nullptr, k_bf, N_NODES);
    k_gemm_bf16<DMODEL, DMODEL, 0><<<ggrid(DMODEL), B256, 0, stream>>>(
        h_bf, wb_v, bv, nullptr, v_bf, N_NODES);

    k_zero_f32<<<(N_NODES * DMODEL + 255) / 256, B256, 0, stream>>>(wv_f, N_NODES * DMODEL);
    k_zero_f32<<<(N_NODES * NHEAD + 255) / 256, B256, 0, stream>>>(zn_f, N_NODES * NHEAD);

    k_edge_attn<<<(N_EDGES * 32 + 255) / 256, B256, 0, stream>>>(
        src, dst, q_bf, k_bf, v_bf, wv_f, zn_f);

    k_attn_combine<<<N_NODES, B32, 0, stream>>>(wv_f, zn_f, iwi_f, a_bf);

    // o = a @ Wo.T + bo   (writes into wv_f, which is now free)
    k_gemm_bf16<DMODEL, DMODEL, 0><<<ggrid(DMODEL), B256, 0, stream>>>(
        a_bf, wb_o, bo, wv_f, nullptr, N_NODES);

    k_add_ln<<<N_NODES, B32, 0, stream>>>(h_f, wv_f, ln1_g, ln1_b, t_f, t_bf);

    // f1 = relu(t @ Wff1.T + b)  [N,256] bf16 (reuses x_bf storage)
    k_gemm_bf16<DMODEL, 2 * DMODEL, 1><<<ggrid(2 * DMODEL), B256, 0, stream>>>(
        t_bf, wb_ff1, b_ff1, nullptr, f1_bf, N_NODES);
    // f2 = f1 @ Wff2.T + b  -> fp32 (reuses wv_f)
    k_gemm_bf16<2 * DMODEL, DMODEL, 0><<<ggrid(DMODEL), B256, 0, stream>>>(
        f1_bf, wb_ff2, b_ff2, wv_f, nullptr, N_NODES);

    k_add_ln<<<N_NODES, B32, 0, stream>>>(t_f, wv_f, ln2_g, ln2_b, h_f, h_bf);
  }

  // outputs: h, then x_hat = selu(h @ Wm1.T + b) @ Wm2.T + b
  k_copy_f32<<<(N_NODES * DMODEL + 255) / 256, B256, 0, stream>>>(h_f, out_h, N_NODES * DMODEL);
  k_gemm_bf16<DMODEL, DMODEL, 2><<<ggrid(DMODEL), B256, 0, stream>>>(
      h_bf, wb_m1, b_m1, nullptr, a_bf, N_NODES);
  k_gemm_bf16<DMODEL, IN_DIM, 0><<<ggrid(IN_DIM), B256, 0, stream>>>(
      a_bf, wb_m2, b_m2, out_x, nullptr, N_NODES);
}
